// CharBiLSTMEmbedder_8744553415088
// MI455X (gfx1250) — matrix operand order repaired and checked by reading, not compiled, not run on hardware
//
#include <hip/hip_runtime.h>
#include <hip/hip_bf16.h>
#include <stdint.h>

#define T_MAXX      16
#define E_DIMM      128
#define H_DIMM      256
#define SEQ_TILE    16
#define BLOCK_THREADS 512   // 16 waves: 8 fwd + 8 bwd

#define N_IH (4 * H_DIMM * E_DIMM)   // 131072 elements
#define N_HH (4 * H_DIMM * H_DIMM)   // 262144 elements

typedef __attribute__((ext_vector_type(16))) __bf16 v16bf;
typedef __attribute__((ext_vector_type(8)))  float  v8f;

union Frag16 { v16bf v; uint4 q[2]; };

__device__ __forceinline__ unsigned short f2bf(float f) {
    unsigned int u = __float_as_uint(f);
    u += 0x7FFFu + ((u >> 16) & 1u);          // round-to-nearest-even
    return (unsigned short)(u >> 16);
}
__device__ __forceinline__ float sigmoid_fast(float x) {
    return 1.0f / (1.0f + __expf(-x));
}
__device__ __forceinline__ float tanh_fast(float x) {
    return 2.0f / (1.0f + __expf(-2.0f * x)) - 1.0f;
}

// ---- pack f32 weights -> bf16 (row-major preserved: W[n][k]) ----
__global__ void bf16_pack_kernel(const float* __restrict__ src,
                                 unsigned short* __restrict__ dst, int n) {
    int i = blockIdx.x * blockDim.x + threadIdx.x;
    if (i < n) dst[i] = f2bf(src[i]);
}

// stage one 16x128 bf16 x_t tile into LDS (256 threads of one direction)
__device__ __forceinline__ void stage_x(unsigned short* __restrict__ xs,
                                        const int* __restrict__ char_ids,
                                        const float* __restrict__ emb,
                                        int seq0, int t, int lt) {
    for (int i = lt; i < SEQ_TILE * E_DIMM; i += 256) {
        int m = i >> 7;               // E_DIMM == 128
        int k = i & (E_DIMM - 1);
        int id = char_ids[(seq0 + m) * T_MAXX + t];
        xs[m * E_DIMM + k] = f2bf(emb[id * E_DIMM + k]);
    }
}

__global__ __launch_bounds__(BLOCK_THREADS)
void bilstm_wmma_kernel(const int*   __restrict__ char_ids,
                        const int*   __restrict__ lengths,
                        const float* __restrict__ emb,
                        const unsigned short* __restrict__ wts,  // packed bf16 weights
                        const float* __restrict__ bF,
                        const float* __restrict__ bB,
                        float* __restrict__ out)
{
    __shared__ unsigned short Xs[2][SEQ_TILE * E_DIMM];   // 4 KB per dir
    __shared__ unsigned short Hs[2][SEQ_TILE * H_DIMM];   // 8 KB per dir

    const int tid   = threadIdx.x;
    const int lane  = tid & 31;
    const int wave  = tid >> 5;
    const int dir   = wave >> 3;        // 0 = fwd, 1 = bwd
    const int wslab = wave & 7;         // 32-unit hidden slab
    const int half  = lane >> 4;
    const int nl    = lane & 15;
    const int lt    = tid & 255;        // lane id within direction half
    const int seq0  = blockIdx.x * SEQ_TILE;

    const float* bias = dir ? bB : bF;
    // element offsets of this direction's weight matrices inside wts[]
    const int wih_off = dir ? (N_IH + N_HH) : 0;
    const int whh_off = dir ? (2 * N_IH + N_HH) : N_IH;

    // per-lane constants -----------------------------------------------------
    int lenv[8];
    #pragma unroll
    for (int r = 0; r < 8; ++r) lenv[r] = lengths[seq0 + half * 8 + r];

    int   nb[8];
    float biasv[8];
    int   voff_ih[8], voff_hh[8];       // 32-bit byte offsets (GVS addressing)
    #pragma unroll
    for (int tt = 0; tt < 8; ++tt) {
        int gate = tt >> 1;                                 // i,f,g,o
        nb[tt]   = gate * H_DIMM + wslab * 32 + (tt & 1) * 16;
        biasv[tt] = bias[nb[tt] + nl];
        voff_ih[tt] = 2 * (wih_off + (nb[tt] + nl) * E_DIMM + half * 16);
        voff_hh[tt] = 2 * (whh_off + (nb[tt] + nl) * H_DIMM + half * 16);
        __builtin_prefetch((const char*)wts + voff_ih[tt], 0, 3);
        __builtin_prefetch((const char*)wts + voff_hh[tt], 0, 3);
    }

    float cst[16], hst[16];
    #pragma unroll
    for (int i = 0; i < 16; ++i) { cst[i] = 0.0f; hst[i] = 0.0f; }

    // prologue: stage x for first step of this direction, zero h ------------
    stage_x(Xs[dir], char_ids, emb, seq0, dir ? (T_MAXX - 1) : 0, lt);
    for (int i = tid; i < 2 * SEQ_TILE * H_DIMM; i += BLOCK_THREADS)
        (&Hs[0][0])[i] = 0;
    __syncthreads();

    const char* wb = (const char*)wts;   // uniform base -> SGPR pair (saddr)

    #pragma unroll 1
    for (int s = 0; s < T_MAXX; ++s) {
        const int tcur = dir ? (T_MAXX - 1 - s) : s;

        // Launder the 32-bit weight offsets through VGPRs each timestep so
        // the B-fragment loads are not provably loop-invariant (prevents
        // hoist-all-96-fragments + scratch spill). Loads re-issue each step
        // and hit WGP$/L2 (weights are 1.5 MB bf16, L2-resident).
        #pragma unroll
        for (int tt = 0; tt < 8; ++tt)
            asm volatile("" : "+v"(voff_ih[tt]), "+v"(voff_hh[tt]));

        v8f acc[8];
        #pragma unroll
        for (int tt = 0; tt < 8; ++tt) {
            #pragma unroll
            for (int e = 0; e < 8; ++e) acc[tt][e] = biasv[tt];
        }

        // ---- input projection: A = x_t [16x128], B = Wih^T [128x1024] ----
        const unsigned short* xsl = &Xs[dir][0];
        #pragma unroll
        for (int kc = 0; kc < E_DIMM; kc += 32) {
            Frag16 a;
            const unsigned short* p0 = xsl + nl * E_DIMM + kc + half * 16;
            a.q[0] = *(const uint4*)p0;
            a.q[1] = *(const uint4*)(p0 + 8);
            Frag16 b[8];
            #pragma unroll
            for (int tt = 0; tt < 8; ++tt) {   // saddr + voffset + imm(kc*2)
                const uint4* bp = (const uint4*)(wb + voff_ih[tt] + kc * 2);
                b[tt].q[0] = bp[0]; b[tt].q[1] = bp[1];
            }
            #pragma unroll
            for (int tt = 0; tt < 8; ++tt) {
                acc[tt] = __builtin_amdgcn_wmma_f32_16x16x32_bf16(
                    false, a.v, false, b[tt].v, (short)0, acc[tt], false, false);
            }
        }

        // ---- recurrent projection: A = h [16x256], B = Whh^T [256x1024] --
        const unsigned short* hsl = &Hs[dir][0];
        #pragma unroll
        for (int kc = 0; kc < H_DIMM; kc += 32) {
            Frag16 a;
            const unsigned short* p0 = hsl + nl * H_DIMM + kc + half * 16;
            a.q[0] = *(const uint4*)p0;
            a.q[1] = *(const uint4*)(p0 + 8);
            Frag16 b[8];
            #pragma unroll
            for (int tt = 0; tt < 8; ++tt) {
                const uint4* bp = (const uint4*)(wb + voff_hh[tt] + kc * 2);
                b[tt].q[0] = bp[0]; b[tt].q[1] = bp[1];
            }
            #pragma unroll
            for (int tt = 0; tt < 8; ++tt) {
                acc[tt] = __builtin_amdgcn_wmma_f32_16x16x32_bf16(
                    false, a.v, false, b[tt].v, (short)0, acc[tt], false, false);
            }
        }

        __syncthreads();   // all LDS reads of x_t / h_prev complete

        // ---- lane-local LSTM cell update + write new h to LDS ------------
        #pragma unroll
        for (int hh = 0; hh < 2; ++hh) {
            #pragma unroll
            for (int r = 0; r < 8; ++r) {
                float iv = sigmoid_fast(acc[0 + hh][r]);
                float fv = sigmoid_fast(acc[2 + hh][r]);
                float gv = tanh_fast  (acc[4 + hh][r]);
                float ov = sigmoid_fast(acc[6 + hh][r]);
                int   idx = hh * 8 + r;
                float cn = fv * cst[idx] + iv * gv;
                float hn = ov * tanh_fast(cn);
                bool upd = tcur < lenv[r];
                cst[idx] = upd ? cn : cst[idx];
                hst[idx] = upd ? hn : hst[idx];
                int m = r + 8 * half;
                int j = wslab * 32 + hh * 16 + nl;
                Hs[dir][m * H_DIMM + j] = f2bf(hst[idx]);
            }
        }
        // stage next x_t for this direction (buffer reads finished above)
        if (s + 1 < T_MAXX)
            stage_x(Xs[dir], char_ids, emb, seq0, dir ? (T_MAXX - 2 - s) : (s + 1), lt);

        __syncthreads();   // new h / new x visible for next step
    }

    // ---- output: [N, 512] = concat(h_fwd, h_bwd), f32 ---------------------
    #pragma unroll
    for (int hh = 0; hh < 2; ++hh) {
        #pragma unroll
        for (int r = 0; r < 8; ++r) {
            int m = r + 8 * half;
            int j = wslab * 32 + hh * 16 + nl;
            out[(size_t)(seq0 + m) * (2 * H_DIMM) + dir * H_DIMM + j] = hst[hh * 8 + r];
        }
    }
}

extern "C" void kernel_launch(void* const* d_in, const int* in_sizes, int n_in,
                              void* d_out, int out_size, void* d_ws, size_t ws_size,
                              hipStream_t stream) {
    const int*   char_ids = (const int*)  d_in[0];
    const int*   lengths  = (const int*)  d_in[1];
    const float* emb      = (const float*)d_in[2];
    const float* Wih_f    = (const float*)d_in[3];
    const float* Whh_f    = (const float*)d_in[4];
    const float* b_f      = (const float*)d_in[5];
    const float* Wih_b    = (const float*)d_in[6];
    const float* Whh_b    = (const float*)d_in[7];
    const float* b_b      = (const float*)d_in[8];
    float* out = (float*)d_out;

    // packed bf16 weights, single allocation: [WihF | WhhF | WihB | WhhB]
    unsigned short* ws   = (unsigned short*)d_ws;
    unsigned short* WihF = ws;
    unsigned short* WhhF = WihF + N_IH;
    unsigned short* WihB = WhhF + N_HH;
    unsigned short* WhhB = WihB + N_IH;     // total 1.5 MB of d_ws

    bf16_pack_kernel<<<(N_IH + 255) / 256, 256, 0, stream>>>(Wih_f, WihF, N_IH);
    bf16_pack_kernel<<<(N_HH + 255) / 256, 256, 0, stream>>>(Whh_f, WhhF, N_HH);
    bf16_pack_kernel<<<(N_IH + 255) / 256, 256, 0, stream>>>(Wih_b, WihB, N_IH);
    bf16_pack_kernel<<<(N_HH + 255) / 256, 256, 0, stream>>>(Whh_b, WhhB, N_HH);

    const int n_words = in_sizes[1];                 // 16384
    const int nblocks = n_words / SEQ_TILE;          // 1024 workgroups
    bilstm_wmma_kernel<<<nblocks, BLOCK_THREADS, 0, stream>>>(
        char_ids, lengths, emb, ws, b_f, b_b, out);
}
